// GLSTMEncoder_68848325755452
// MI455X (gfx1250) — matrix operand order repaired and checked by reading, not compiled, not run on hardware
//
#include <hip/hip_runtime.h>

#define DIN  256
#define DOUT 128
#define TRI  384
#define LVLS 9

typedef float  v8f  __attribute__((ext_vector_type(8)));
typedef __bf16 v16bf __attribute__((ext_vector_type(16)));
typedef __bf16 v8bf  __attribute__((ext_vector_type(8)));

union Frag16 { v16bf v; v8bf h[2]; };

__device__ __forceinline__ float sigf(float x) { return 1.0f / (1.0f + __expf(-x)); }

// round-to-nearest-even f32 -> bf16 (as raw u16), and back
__device__ __forceinline__ unsigned short bf16_rne(float f) {
    unsigned u = __float_as_uint(f);
    unsigned r = 0x7FFFu + ((u >> 16) & 1u);
    return (unsigned short)((u + r) >> 16);
}
__device__ __forceinline__ float bf16_f32(unsigned short h) {
    return __uint_as_float(((unsigned)h) << 16);
}
__device__ __forceinline__ void split2(float f, unsigned short* hi, unsigned short* lo) {
    unsigned short h = bf16_rne(f);
    *hi = h;
    *lo = bf16_rne(f - bf16_f32(h));
}

// ---------------- setup kernels ----------------

__global__ void zero_i32_kernel(int* __restrict__ p, int n) {
    int i = blockIdx.x * blockDim.x + threadIdx.x;
    if (i < n) p[i] = 0;
}

// f32 -> (bf16 hi, bf16 lo-residual) split for the WMMA operands
__global__ void split_kernel(const float* __restrict__ s, unsigned short* __restrict__ hi,
                             unsigned short* __restrict__ lo, long n) {
    long i = (long)blockIdx.x * blockDim.x + threadIdx.x;
    if (i < n) split2(s[i], &hi[i], &lo[i]);
}

__global__ void degree_kernel(const int* __restrict__ src, int* __restrict__ order, int E) {
    int e = blockIdx.x * blockDim.x + threadIdx.x;
    if (e < E) atomicAdd(&order[src[e]], 1);
}

// edges are sorted by parent (src); detect row boundaries -> CSR row starts
__global__ void boundary_kernel(const int* __restrict__ src, int* __restrict__ rowstart, int E) {
    int e = blockIdx.x * blockDim.x + threadIdx.x;
    if (e < E) {
        if (e == 0 || src[e] != src[e - 1]) rowstart[src[e]] = e;
    }
}

__global__ void classify_kernel(const int* __restrict__ order, int* __restrict__ active,
                                int* __restrict__ counts, int N) {
    int n = blockIdx.x * blockDim.x + threadIdx.x;
    if (n < N) {
        int l = order[n];
        if (l < LVLS) {
            int pos = atomicAdd(&counts[l], 1);
            active[(size_t)l * N + pos] = n;
        }
    }
}

// ---------------- bf16x3 WMMA GEMM (fp32-accurate via hi/lo split) ----------------
// C[rowC[m], n] (+)= sum_k A[m,k]*W[n,k] + bias[n], A = Ahi+Alo, W = Whi+Wlo (bf16 splits)
// product = Ahi*Whi + Alo*Whi + Ahi*Wlo accumulated in f32 (v_wmma_f32_16x16x32_bf16).
// 1 wave/block: 32 rows (2 M-tiles) x 64 cols (4 N-tiles). K template const, mult of 32.
// Addressing: uniform base pointers + 6 shared 32-bit lane offsets; k0 folds into the
// load immediate (full unroll) -> no address ALU and no spills in the hot loop.
template <int K>
__global__ __launch_bounds__(32) void gemm_wmma_bf16x3(
        const unsigned short* __restrict__ Ahi,
        const unsigned short* __restrict__ Alo,
        const unsigned short* __restrict__ Whi,  // [Nout,K] row-major
        const unsigned short* __restrict__ Wlo,
        const float* __restrict__ bias,          // [Nout] or null
        float* __restrict__ C,
        const int* __restrict__ rowC,            // null => identity
        const int* __restrict__ cntPtr,          // null => use M
        int M, int ldc, int accumulate) {
    const int lane = threadIdx.x & 31;
    int Mc = M;
    if (cntPtr) { int cv = *cntPtr; Mc = cv < M ? cv : M; }
    const int m0 = blockIdx.x * 32;
    if (m0 >= Mc) return;                   // wave-uniform: EXEC stays all-ones
    const int n0    = blockIdx.y * 64;
    const int half  = lane >> 4;            // 0/1
    const int mlane = lane & 15;

    // A 16x32 layout (matches B): half 0 holds K [0..15], half 1 [16..31]
    unsigned aOff[2];                       // element offsets shared by Ahi/Alo
#pragma unroll
    for (int mt = 0; mt < 2; ++mt) {
        int mr = m0 + mt * 16 + mlane;
        unsigned r = (mr < Mc) ? (unsigned)mr : 0u;   // dummy row 0, stores guarded below
        aOff[mt] = r * (unsigned)K + (unsigned)(16 * half);
    }
    // B = W^T 32x16: lane n holds contiguous K [16*half, 16*half+16) of W row n
    unsigned bOff[4];                       // element offsets shared by Whi/Wlo
#pragma unroll
    for (int t = 0; t < 4; ++t)
        bOff[t] = (unsigned)(n0 + t * 16 + mlane) * (unsigned)K + (unsigned)(16 * half);

    v8f acc[2][4] = {};
#pragma unroll
    for (int k0 = 0; k0 < K; k0 += 32) {
        Frag16 ah[2], al[2];
#pragma unroll
        for (int mt = 0; mt < 2; ++mt) {
            ah[mt].h[0] = *(const v8bf*)(Ahi + aOff[mt] + k0);
            ah[mt].h[1] = *(const v8bf*)(Ahi + aOff[mt] + k0 + 8);
            al[mt].h[0] = *(const v8bf*)(Alo + aOff[mt] + k0);
            al[mt].h[1] = *(const v8bf*)(Alo + aOff[mt] + k0 + 8);
        }
#pragma unroll
        for (int t = 0; t < 4; ++t) {
            v16bf bh = *(const v16bf*)(Whi + bOff[t] + k0);
            v16bf bl = *(const v16bf*)(Wlo + bOff[t] + k0);
#pragma unroll
            for (int mt = 0; mt < 2; ++mt) {
                acc[mt][t] = __builtin_amdgcn_wmma_f32_16x16x32_bf16(
                    false, ah[mt].v, false, bh, (short)0, acc[mt][t], false, false);
                acc[mt][t] = __builtin_amdgcn_wmma_f32_16x16x32_bf16(
                    false, al[mt].v, false, bh, (short)0, acc[mt][t], false, false);
                acc[mt][t] = __builtin_amdgcn_wmma_f32_16x16x32_bf16(
                    false, ah[mt].v, false, bl, (short)0, acc[mt][t], false, false);
            }
        }
    }

    // C/D layout: VGPR r holds (M = r + 8*half, N = lane&15)
#pragma unroll
    for (int mt = 0; mt < 2; ++mt) {
#pragma unroll
        for (int r = 0; r < 8; ++r) {
            int m = m0 + mt * 16 + r + half * 8;
            if (m >= Mc) continue;
            int row = rowC ? rowC[m] : m;
            float* Cp = C + (size_t)row * ldc + n0 + mlane;
#pragma unroll
            for (int t = 0; t < 4; ++t) {
                int n = n0 + t * 16 + mlane;
                float v = acc[mt][t][r];
                if (bias) v += bias[n];
                if (accumulate) v += Cp[t * 16];
                Cp[t * 16] = v;
            }
        }
    }
}

// ---------------- per-level kernels ----------------

// compact child-sum rows, emitted directly as bf16 hi/lo split for the U_iou GEMM
__global__ void hsum_gather_kernel(const int* __restrict__ active, const int* __restrict__ cnt,
                                   const int* __restrict__ rowstart, const int* __restrict__ dst,
                                   const float* __restrict__ h,
                                   unsigned short* __restrict__ hsH, unsigned short* __restrict__ hsL,
                                   int lvl, int MB) {
    int tid = blockIdx.x * blockDim.x + threadIdx.x;
    int m = tid >> 7, j = tid & 127;
    int n = *cnt; if (n > MB) n = MB;
    if (m >= n) return;
    int p  = active[m];
    int e0 = rowstart[p];          // in-degree of a level-lvl node == lvl
    float s = 0.0f;
    for (int q = 0; q < lvl; ++q)
        s += h[(size_t)dst[e0 + q] * DOUT + j];
    split2(s, &hsH[(size_t)m * DOUT + j], &hsL[(size_t)m * DOUT + j]);
}

// fused: c = sig(i)*tanh(u); c += sum_children sig(wx_f + uh[d])*c[d]; h = sig(o)*tanh(c)
// also emits compact bf16 hi/lo h rows for this level's U_f GEMM
__global__ void level_node_kernel(const int* __restrict__ active, const int* __restrict__ cnt,
                                  const int* __restrict__ rowstart, const int* __restrict__ dst,
                                  const float* __restrict__ wx_iou, const float* __restrict__ wx_f,
                                  const float* __restrict__ uh, float* __restrict__ c,
                                  float* __restrict__ h,
                                  unsigned short* __restrict__ hH, unsigned short* __restrict__ hL,
                                  int lvl, int MB) {
    int tid = blockIdx.x * blockDim.x + threadIdx.x;
    int m = tid >> 7, j = tid & 127;
    int n = *cnt; if (n > MB) n = MB;
    if (m >= n) return;
    int p = active[m];
    const float* iou = wx_iou + (size_t)p * TRI;
    float iv = sigf(iou[j]);
    float ov = sigf(iou[DOUT + j]);
    float uv = tanhf(iou[2 * DOUT + j]);
    float cv = iv * uv;
    if (lvl > 0) {
        int e0 = rowstart[p];
        float wf = wx_f[(size_t)p * DOUT + j];
        for (int q = 0; q < lvl; ++q) {
            int d = dst[e0 + q];
            cv += sigf(wf + uh[(size_t)d * DOUT + j]) * c[(size_t)d * DOUT + j];
        }
    }
    float hv = ov * tanhf(cv);
    c[(size_t)p * DOUT + j] = cv;
    h[(size_t)p * DOUT + j] = hv;
    split2(hv, &hH[(size_t)m * DOUT + j], &hL[(size_t)m * DOUT + j]);
}

// ---------------- launch ----------------

extern "C" void kernel_launch(void* const* d_in, const int* in_sizes, int n_in,
                              void* d_out, int out_size, void* d_ws, size_t ws_size,
                              hipStream_t stream) {
    const float* x       = (const float*)d_in[0];
    const float* W_iou_w = (const float*)d_in[1];
    const float* W_iou_b = (const float*)d_in[2];
    const float* U_iou_w = (const float*)d_in[3];
    const float* U_iou_b = (const float*)d_in[4];
    const float* W_f_w   = (const float*)d_in[5];
    const float* W_f_b   = (const float*)d_in[6];
    const float* U_f_w   = (const float*)d_in[7];
    const float* U_f_b   = (const float*)d_in[8];
    const int*   edge    = (const int*)d_in[9];

    const int N = in_sizes[0] / DIN;
    const int E = in_sizes[9] / 2;
    const int* src = edge;
    const int* dst = edge + E;

    float* h = (float*)d_out;                  // [N, 128] output
    const int MB = N / LVLS + 64;              // per-level active-count upper bound

    // workspace carve-out (256B-aligned so bf16 vector loads are 32B-aligned)
    char* base = (char*)d_ws;
    size_t off = 0;
    auto alloc = [&](size_t bytes) -> void* {
        void* q = base + off;
        off = (off + bytes + 255) & ~(size_t)255;
        return q;
    };
    float* wx_iou = (float*)alloc((size_t)N * TRI * 4);   // x@W_iou^T, += hsum@U_iou^T in place
    float* wx_f   = (float*)alloc((size_t)N * DOUT * 4);
    float* c      = (float*)alloc((size_t)N * DOUT * 4);
    float* uh     = (float*)alloc((size_t)N * DOUT * 4);  // h@U_f^T, once per node at its level
    unsigned short* xH  = (unsigned short*)alloc((size_t)N * DIN * 2);
    unsigned short* xL  = (unsigned short*)alloc((size_t)N * DIN * 2);
    unsigned short* WiH = (unsigned short*)alloc((size_t)TRI * DIN * 2);
    unsigned short* WiL = (unsigned short*)alloc((size_t)TRI * DIN * 2);
    unsigned short* WfH = (unsigned short*)alloc((size_t)DOUT * DIN * 2);
    unsigned short* WfL = (unsigned short*)alloc((size_t)DOUT * DIN * 2);
    unsigned short* UiH = (unsigned short*)alloc((size_t)TRI * DOUT * 2);
    unsigned short* UiL = (unsigned short*)alloc((size_t)TRI * DOUT * 2);
    unsigned short* UfH = (unsigned short*)alloc((size_t)DOUT * DOUT * 2);
    unsigned short* UfL = (unsigned short*)alloc((size_t)DOUT * DOUT * 2);
    unsigned short* hsH = (unsigned short*)alloc((size_t)MB * DOUT * 2);  // compact splits
    unsigned short* hsL = (unsigned short*)alloc((size_t)MB * DOUT * 2);
    unsigned short* hH  = (unsigned short*)alloc((size_t)MB * DOUT * 2);
    unsigned short* hL  = (unsigned short*)alloc((size_t)MB * DOUT * 2);
    int* order    = (int*)alloc((size_t)N * 4);
    int* rowstart = (int*)alloc((size_t)N * 4);
    int* active   = (int*)alloc((size_t)LVLS * N * 4);
    int* counts   = (int*)alloc(64);

    auto cdiv = [](long a, long b) { return (int)((a + b - 1) / b); };

    // phase A: topology, operand splits, big GEMMs
    zero_i32_kernel<<<cdiv(N, 256), 256, 0, stream>>>(order, N);
    zero_i32_kernel<<<1, 32, 0, stream>>>(counts, 16);
    degree_kernel<<<cdiv(E, 256), 256, 0, stream>>>(src, order, E);
    boundary_kernel<<<cdiv(E, 256), 256, 0, stream>>>(src, rowstart, E);
    classify_kernel<<<cdiv(N, 256), 256, 0, stream>>>(order, active, counts, N);

    split_kernel<<<cdiv((long)N * DIN, 256), 256, 0, stream>>>(x, xH, xL, (long)N * DIN);
    split_kernel<<<cdiv((long)TRI * DIN, 256), 256, 0, stream>>>(W_iou_w, WiH, WiL, (long)TRI * DIN);
    split_kernel<<<cdiv((long)DOUT * DIN, 256), 256, 0, stream>>>(W_f_w, WfH, WfL, (long)DOUT * DIN);
    split_kernel<<<cdiv((long)TRI * DOUT, 256), 256, 0, stream>>>(U_iou_w, UiH, UiL, (long)TRI * DOUT);
    split_kernel<<<cdiv((long)DOUT * DOUT, 256), 256, 0, stream>>>(U_f_w, UfH, UfL, (long)DOUT * DOUT);

    gemm_wmma_bf16x3<DIN><<<dim3(cdiv(N, 32), TRI / 64), 32, 0, stream>>>(
        xH, xL, WiH, WiL, W_iou_b, wx_iou, nullptr, nullptr, N, TRI, 0);
    gemm_wmma_bf16x3<DIN><<<dim3(cdiv(N, 32), DOUT / 64), 32, 0, stream>>>(
        xH, xL, WfH, WfL, W_f_b, wx_f, nullptr, nullptr, N, DOUT, 0);

    // phase B: 9 dependency levels
    const int nodeThreads = MB * DOUT;
    for (int lvl = 0; lvl < LVLS; ++lvl) {
        const int* act = active + (size_t)lvl * N;
        const int* cnt = counts + lvl;
        if (lvl > 0) {
            hsum_gather_kernel<<<cdiv(nodeThreads, 256), 256, 0, stream>>>(
                act, cnt, rowstart, dst, h, hsH, hsL, lvl, MB);
            // wx_iou[p,:] += hsum @ U_iou^T + U_iou_b  (each node active exactly once)
            gemm_wmma_bf16x3<DOUT><<<dim3(cdiv(MB, 32), TRI / 64), 32, 0, stream>>>(
                hsH, hsL, UiH, UiL, U_iou_b, wx_iou, act, cnt, MB, TRI, 1);
        }
        level_node_kernel<<<cdiv(nodeThreads, 256), 256, 0, stream>>>(
            act, cnt, rowstart, dst, wx_iou, wx_f, uh, c, h, hH, hL, lvl, MB);
        if (lvl < LVLS - 1) {
            // uh[p,:] = h[p,:] @ U_f^T + U_f_b for just-finalized nodes
            gemm_wmma_bf16x3<DOUT><<<dim3(cdiv(MB, 32), DOUT / 64), 32, 0, stream>>>(
                hH, hL, UfH, UfL, U_f_b, uh, act, cnt, MB, DOUT, 0);
        }
    }
}
